// MultiheadAttention_76089640615944
// MI455X (gfx1250) — matrix operand order repaired and checked by reading, not compiled, not run on hardware
//
#include <hip/hip_runtime.h>
#include <hip/hip_bf16.h>

typedef _Float16 f16;
typedef __attribute__((ext_vector_type(16))) _Float16 v16h;
typedef __attribute__((ext_vector_type(8)))  _Float16 v8h;
typedef __attribute__((ext_vector_type(2)))  _Float16 h2;
typedef __attribute__((ext_vector_type(8)))  float    v8f;

#define B_  2
#define L_  2048
#define D_  1024
#define H_  16
#define HD_ 64

// ---------------------------------------------------------------------------
// D = A(16x32 f16) * B(32x16 f16) + C(16x16 f32), wave32 WMMA
// ---------------------------------------------------------------------------
static __device__ __forceinline__ v8f wmma16x16x32(v16h a, v16h b, v8f c) {
  return __builtin_amdgcn_wmma_f32_16x16x32_f16(
      /*neg_a=*/false, a, /*neg_b=*/false, b,
      /*c_mod=*/(short)0, c, /*reuse_a=*/false, /*reuse_b=*/false);
}

static __device__ __forceinline__ int h2_to_i(h2 v) { union { h2 h; int i; } u; u.h = v; return u.i; }
static __device__ __forceinline__ h2  i_to_h2(int v) { union { h2 h; int i; } u; u.i = v; return u.h; }
static __device__ __forceinline__ h2  pkmax(h2 a, h2 b) {
  h2 r; r[0] = a[0] > b[0] ? a[0] : b[0]; r[1] = a[1] > b[1] ? a[1] : b[1]; return r;
}

// A fragment 16x32 (MxK), f16 row-major source with leading dim ld.
// Contiguous k order (B uses the same): lane<16 -> row=lane, K = 0..15; lane>=16 -> row=lane-16,
// K = 16..31.
static __device__ __forceinline__ v16h load_a_f16(const f16* __restrict__ src,
                                                  int ld, int m0, int k0, int lane) {
  const int half = lane >> 4;
  const f16* p = src + (size_t)(m0 + (lane & 15)) * ld + k0 + 16 * half;
  v8h lo = *(const v8h*)(p);
  v8h hi = *(const v8h*)(p + 8);
  v16h a;
#pragma unroll
  for (int i = 0; i < 8; ++i) { a[i] = lo[i]; a[8 + i] = hi[i]; }
  return a;
}

// B fragment 32x16 (KxN) where B[k][n] = src[n0+n][k0+k] (src row-major over k,
// leading dim ld). Contiguous k order (matches A): lane<16 -> k=0..15, lane>=16 -> k=16..31; n=lane&15.
// Per lane this is one contiguous 32-byte f16 load.
static __device__ __forceinline__ v16h load_bT_f16(const f16* __restrict__ src,
                                                   int ld, int n0, int k0, int lane) {
  const f16* p = src + (size_t)(n0 + (lane & 15)) * ld + k0 + 16 * (lane >> 4);
  return *(const v16h*)p;
}

// ---------------------------------------------------------------------------
// Kernel 0: one-shot fp32 -> f16 conversion (x and the four weights).
// Pure streaming: ~32 MB in / 16 MB out, ~2 us at 23.3 TB/s.
// ---------------------------------------------------------------------------
__global__ __launch_bounds__(256) void cvt_f32_f16_kernel(
    const float* __restrict__ src, f16* __restrict__ dst, int n8)
{
  const int i = blockIdx.x * 256 + threadIdx.x;
  if (i >= n8) return;
  const float4* s = (const float4*)src + (size_t)i * 2;
  const float4 a = s[0], b = s[1];
  v8h o;
  o[0] = (f16)a.x; o[1] = (f16)a.y; o[2] = (f16)a.z; o[3] = (f16)a.w;
  o[4] = (f16)b.x; o[5] = (f16)b.y; o[6] = (f16)b.z; o[7] = (f16)b.w;
  *(v8h*)(dst + (size_t)i * 8) = o;
}

// ---------------------------------------------------------------------------
// Kernel 1: fused QKV projection.  y = x @ W^T + b, f16 in / f16 out.
// 2x2 register blocking: each wave computes a 32x32 output patch.
// grid = (D/32, (B*L)/(32*8), 3) ; block = 256 (8 waves).
// z=0 -> Q (per-(b,h) L x 64), z=1 -> K (same), z=2 -> V^T (per-(b,h) 64 x L)
// ---------------------------------------------------------------------------
__global__ __launch_bounds__(256) void qkv_proj_kernel(
    const f16* __restrict__ x16, const f16* __restrict__ w16,
    const float* __restrict__ bq, const float* __restrict__ bk,
    const float* __restrict__ bv,
    f16* __restrict__ q_ws, f16* __restrict__ k_ws, f16* __restrict__ vt_ws)
{
  const int lane = threadIdx.x & 31;
  const int wave = threadIdx.x >> 5;
  const int n0 = blockIdx.x * 32;
  const int m0 = (blockIdx.y * 8 + wave) * 32;
  const int z  = blockIdx.z;

  const f16*   w  = w16 + (size_t)z * D_ * D_;
  const float* bi = (z == 0) ? bq : (z == 1) ? bk : bv;

  v8f acc[2][2] = {};
#pragma unroll 2
  for (int k0 = 0; k0 < D_; k0 += 32) {
    if (k0 + 32 < D_) {
      __builtin_prefetch(w   + (size_t)(n0 + (lane & 15)) * D_ + k0 + 32, 0, 1);
      __builtin_prefetch(x16 + (size_t)(m0 + (lane & 15)) * D_ + k0 + 32, 0, 1);
    }
    const v16h a0 = load_a_f16(x16, D_, m0,      k0, lane);
    const v16h a1 = load_a_f16(x16, D_, m0 + 16, k0, lane);
    const v16h b0 = load_bT_f16(w, D_, n0,      k0, lane);
    const v16h b1 = load_bT_f16(w, D_, n0 + 16, k0, lane);
    acc[0][0] = wmma16x16x32(a0, b0, acc[0][0]);
    acc[0][1] = wmma16x16x32(a0, b1, acc[0][1]);
    acc[1][0] = wmma16x16x32(a1, b0, acc[1][0]);
    acc[1][1] = wmma16x16x32(a1, b1, acc[1][1]);
  }

  const int half = lane >> 4;
#pragma unroll
  for (int nt = 0; nt < 2; ++nt) {
    const int n = n0 + 16 * nt + (lane & 15);
    const float bias = bi[n];
    const int h  = n >> 6;   // head
    const int hd = n & 63;   // dim within head
#pragma unroll
    for (int mt = 0; mt < 2; ++mt) {
      if (z < 2) {
        f16* dst = (z == 0) ? q_ws : k_ws;
#pragma unroll
        for (int r = 0; r < 8; ++r) {
          const int m  = m0 + 16 * mt + 8 * half + r;  // global token row
          const int b_ = m >> 11;
          const int l  = m & (L_ - 1);
          dst[(((size_t)(b_ * H_ + h)) * L_ + l) * HD_ + hd] =
              (f16)(acc[mt][nt][r] + bias);
        }
      } else {
        // V transposed: vt[((b*H+h)*HD+hd)*L + l]; 8 consecutive l -> b128 store
        const int mbase = m0 + 16 * mt + 8 * half;
        const int b_ = mbase >> 11;
        const int l  = mbase & (L_ - 1);
        v8h pk;
#pragma unroll
        for (int r = 0; r < 8; ++r) pk[r] = (f16)(acc[mt][nt][r] + bias);
        *(v8h*)(vt_ws + (((size_t)(b_ * H_ + h)) * HD_ + hd) * L_ + l) = pk;
      }
    }
  }
}

// ---------------------------------------------------------------------------
// Kernel 2: flash attention.  1 wave = 16 query rows; 8 waves/block.
// grid = (L/128, B*H) ; block = 256.  Online softmax over 32-key blocks.
//  - softmax scale folded into the Q fragment (0.125 exact in f16)
//  - row-sums computed on the matrix pipe via a ones-matrix WMMA (o4)
//  - row-max reduced with packed-f16 shuffles (2 rows per 32-bit register)
// ---------------------------------------------------------------------------
__global__ __launch_bounds__(256) void flash_attn_kernel(
    const f16* __restrict__ q_ws, const f16* __restrict__ k_ws,
    const f16* __restrict__ vt_ws, f16* __restrict__ ao)
{
  __shared__ f16 plds[8 * 16 * 32];   // per-wave 16x32 P staging (8 KB total)

  const int lane = threadIdx.x & 31;
  const int wave = threadIdx.x >> 5;
  const int half = lane >> 4;
  const int bh   = blockIdx.y;
  const int q0   = blockIdx.x * 128 + wave * 16;

  const f16* Q  = q_ws  + (size_t)bh * L_ * HD_;
  const f16* K  = k_ws  + (size_t)bh * L_ * HD_;
  const f16* Vt = vt_ws + (size_t)bh * HD_ * L_;

  // Q fragments, pre-scaled by 1/sqrt(HD) = 0.125 (exact power of two in f16)
  v16h aq0 = load_a_f16(Q, HD_, q0, 0,  lane);
  v16h aq1 = load_a_f16(Q, HD_, q0, 32, lane);
#pragma unroll
  for (int i = 0; i < 16; ++i) { aq0[i] *= (f16)0.125f; aq1[i] *= (f16)0.125f; }

  v16h ones;
#pragma unroll
  for (int i = 0; i < 16; ++i) ones[i] = (f16)1.0f;

  v8f o0 = {}, o1 = {}, o2 = {}, o3 = {};
  v8f o4 = {};                       // online-softmax denominator via P @ 1s
  float mrow[8];
#pragma unroll
  for (int r = 0; r < 8; ++r) mrow[r] = -1e30f;

  f16* pw = plds + wave * (16 * 32);

  for (int kb = 0; kb < L_; kb += 32) {
    const v8f zacc = {};
    // S = (Q*scale) @ K^T for 32 keys: two 16x16 tiles, two k-steps each
    v8f s0 = wmma16x16x32(aq0, load_bT_f16(K, HD_, kb,      0,  lane), zacc);
    s0     = wmma16x16x32(aq1, load_bT_f16(K, HD_, kb,      32, lane), s0);
    v8f s1 = wmma16x16x32(aq0, load_bT_f16(K, HD_, kb + 16, 0,  lane), zacc);
    s1     = wmma16x16x32(aq1, load_bT_f16(K, HD_, kb + 16, 32, lane), s1);

    // Row max over the 32 keys: packed f16, 2 rows per register, 4 shuffle
    // rounds within each 16-lane group (stability max can tolerate f16).
    f16 rmax[8];
#pragma unroll
    for (int rp = 0; rp < 4; ++rp) {
      h2 m2;
      m2[0] = (f16)fmaxf(s0[2 * rp],     s1[2 * rp]);
      m2[1] = (f16)fmaxf(s0[2 * rp + 1], s1[2 * rp + 1]);
#pragma unroll
      for (int off = 1; off < 16; off <<= 1)
        m2 = pkmax(m2, i_to_h2(__shfl_xor(h2_to_i(m2), off, 32)));
      rmax[2 * rp] = m2[0]; rmax[2 * rp + 1] = m2[1];
    }

    // Exponentiate + rescale accumulators.  C/D layout: vgpr r -> row
    // r + 8*half, col = lane&15.
#pragma unroll
    for (int r = 0; r < 8; ++r) {
      const float mn    = fmaxf(mrow[r], (float)rmax[r]);
      const float alpha = __expf(mrow[r] - mn);
      const float p0 = __expf(s0[r] - mn);
      const float p1 = __expf(s1[r] - mn);
      mrow[r] = mn;
      o0[r] *= alpha; o1[r] *= alpha; o2[r] *= alpha; o3[r] *= alpha; o4[r] *= alpha;

      const int row = r + 8 * half;
      const int col = lane & 15;
      pw[row * 32 + col]      = (f16)p0;
      pw[row * 32 + col + 16] = (f16)p1;
    }
    __builtin_amdgcn_wave_barrier();

    // Re-read P as an A-fragment (16x32, keys = contraction dim)
    const f16* pr = pw + (lane & 15) * 32 + 16 * half;
    v8h lo = *(const v8h*)(pr);
    v8h hi = *(const v8h*)(pr + 8);
    v16h ap;
#pragma unroll
    for (int i = 0; i < 8; ++i) { ap[i] = lo[i]; ap[8 + i] = hi[i]; }

    // O += P @ V ; V^T rows contiguous over keys -> single v16h per fragment.
    // o4 += P @ 1s accumulates the softmax denominator on the matrix pipe.
    o0 = wmma16x16x32(ap, load_bT_f16(Vt, L_,  0, kb, lane), o0);
    o1 = wmma16x16x32(ap, load_bT_f16(Vt, L_, 16, kb, lane), o1);
    o2 = wmma16x16x32(ap, load_bT_f16(Vt, L_, 32, kb, lane), o2);
    o3 = wmma16x16x32(ap, load_bT_f16(Vt, L_, 48, kb, lane), o3);
    o4 = wmma16x16x32(ap, ones, o4);
    __builtin_amdgcn_wave_barrier();
  }

  // Normalize and store attention output as f16 in (B, L, D) layout
  const int b_ = bh >> 4;
  const int h  = bh & 15;
#pragma unroll
  for (int r = 0; r < 8; ++r) {
    const float inv = 1.0f / o4[r];        // every column of o4 = row sum
    const int mglob = b_ * L_ + q0 + 8 * half + r;
    const size_t base = (size_t)mglob * D_ + h * HD_ + (lane & 15);
    ao[base +  0] = (f16)(o0[r] * inv);
    ao[base + 16] = (f16)(o1[r] * inv);
    ao[base + 32] = (f16)(o2[r] * inv);
    ao[base + 48] = (f16)(o3[r] * inv);
  }
}

// ---------------------------------------------------------------------------
// Kernel 3: output projection. out = AO @ wo^T + bo (fp32 out), 2x2 blocking.
// grid = (D/32, (B*L)/(32*8)) ; block = 256.
// ---------------------------------------------------------------------------
__global__ __launch_bounds__(256) void out_proj_kernel(
    const f16* __restrict__ ao, const f16* __restrict__ wo16,
    const float* __restrict__ bo, float* __restrict__ out)
{
  const int lane = threadIdx.x & 31;
  const int wave = threadIdx.x >> 5;
  const int n0 = blockIdx.x * 32;
  const int m0 = (blockIdx.y * 8 + wave) * 32;

  v8f acc[2][2] = {};
#pragma unroll 2
  for (int k0 = 0; k0 < D_; k0 += 32) {
    if (k0 + 32 < D_) {
      __builtin_prefetch(wo16 + (size_t)(n0 + (lane & 15)) * D_ + k0 + 32, 0, 1);
      __builtin_prefetch(ao   + (size_t)(m0 + (lane & 15)) * D_ + k0 + 32, 0, 1);
    }
    const v16h a0 = load_a_f16(ao, D_, m0,      k0, lane);
    const v16h a1 = load_a_f16(ao, D_, m0 + 16, k0, lane);
    const v16h b0 = load_bT_f16(wo16, D_, n0,      k0, lane);
    const v16h b1 = load_bT_f16(wo16, D_, n0 + 16, k0, lane);
    acc[0][0] = wmma16x16x32(a0, b0, acc[0][0]);
    acc[0][1] = wmma16x16x32(a0, b1, acc[0][1]);
    acc[1][0] = wmma16x16x32(a1, b0, acc[1][0]);
    acc[1][1] = wmma16x16x32(a1, b1, acc[1][1]);
  }

  const int half = lane >> 4;
#pragma unroll
  for (int nt = 0; nt < 2; ++nt) {
    const int n = n0 + 16 * nt + (lane & 15);
    const float bias = bo[n];
#pragma unroll
    for (int mt = 0; mt < 2; ++mt) {
#pragma unroll
      for (int r = 0; r < 8; ++r)
        out[(size_t)(m0 + 16 * mt + 8 * half + r) * D_ + n] =
            acc[mt][nt][r] + bias;
    }
  }
}

// ---------------------------------------------------------------------------
extern "C" void kernel_launch(void* const* d_in, const int* in_sizes, int n_in,
                              void* d_out, int out_size, void* d_ws, size_t ws_size,
                              hipStream_t stream) {
  (void)in_sizes; (void)n_in; (void)out_size; (void)ws_size;

  const float* x  = (const float*)d_in[0];
  const float* wq = (const float*)d_in[1];
  const float* bq = (const float*)d_in[2];
  const float* wk = (const float*)d_in[3];
  const float* bk = (const float*)d_in[4];
  const float* wv = (const float*)d_in[5];
  const float* bv = (const float*)d_in[6];
  const float* wo = (const float*)d_in[7];
  const float* bo = (const float*)d_in[8];
  float* out = (float*)d_out;

  const size_t tsz = (size_t)B_ * H_ * L_ * HD_;   // 4,194,304 = B*L*D
  const size_t wsz = (size_t)D_ * D_;              // 1,048,576
  f16* q_ws  = (f16*)d_ws;
  f16* k_ws  = q_ws  + tsz;
  f16* vt_ws = k_ws  + tsz;
  f16* ao    = vt_ws + tsz;
  f16* x16   = ao    + tsz;
  f16* w16   = x16   + tsz;        // wq16 | wk16 | wv16 | wo16 (4 * wsz)
  f16* wo16  = w16   + 3 * wsz;    // total scratch = 48 MB

  const dim3 blk(256, 1, 1);

  // one-shot fp32 -> f16 conversions
  cvt_f32_f16_kernel<<<dim3((unsigned)(tsz / 8 / 256)), blk, 0, stream>>>(x, x16, (int)(tsz / 8));
  cvt_f32_f16_kernel<<<dim3((unsigned)(wsz / 8 / 256)), blk, 0, stream>>>(wq, w16 + 0 * wsz, (int)(wsz / 8));
  cvt_f32_f16_kernel<<<dim3((unsigned)(wsz / 8 / 256)), blk, 0, stream>>>(wk, w16 + 1 * wsz, (int)(wsz / 8));
  cvt_f32_f16_kernel<<<dim3((unsigned)(wsz / 8 / 256)), blk, 0, stream>>>(wv, w16 + 2 * wsz, (int)(wsz / 8));
  cvt_f32_f16_kernel<<<dim3((unsigned)(wsz / 8 / 256)), blk, 0, stream>>>(wo, w16 + 3 * wsz, (int)(wsz / 8));

  qkv_proj_kernel<<<dim3(D_ / 32, (B_ * L_) / (32 * 8), 3), blk, 0, stream>>>(
      x16, w16, bq, bk, bv, q_ws, k_ws, vt_ws);
  flash_attn_kernel<<<dim3(L_ / 128, B_ * H_, 1), blk, 0, stream>>>(
      q_ws, k_ws, vt_ws, ao);
  out_proj_kernel<<<dim3(D_ / 32, (B_ * L_) / (32 * 8), 1), blk, 0, stream>>>(
      ao, wo16, bo, out);
}
